// luong_gate_attention_38354057953408
// MI455X (gfx1250) — compile-verified, matrix operand order checked by reading, not run on hardware
//
#include <hip/hip_runtime.h>
#include <cstdint>
#include <cstddef>

// ---------------------------------------------------------------------------
// Luong-gate attention for MI455X (gfx1250, wave32, WMMA).
// context[2048*128, 1024] f32 read once -> bf16 in LDS ->
// v_wmma_f32_16x16x32_bf16 (f32 accum). v spilled bf16 (33.5 MB, L2-resident).
// Wave split: waves 0-3 = K-projection + scores, waves 4-7 = V-projection.
// B fragments register double-buffered to overlap global loads with WMMA.
// ---------------------------------------------------------------------------

typedef __attribute__((ext_vector_type(16))) __bf16 v16bf;
typedef __attribute__((ext_vector_type(8)))  __bf16 v8bf;
typedef __attribute__((ext_vector_type(8)))  float  v8f;

#define B_   128
#define S_   2048
#define H_   512
#define H2_  1024
#define D_   50
#define DP_  64     // D padded to 4 WMMA N-tiles
#define XST_ 136    // LDS row stride in bf16 (128 + 8 pad, keeps 16B align)

__device__ __forceinline__ float selu_f(float x) {
    const float scale = 1.0507009873554804934193349852946f;
    const float alpha = 1.6732632423543772848170429916717f;
    return x > 0.0f ? scale * x : scale * alpha * (__expf(x) - 1.0f);
}

// ---------------------------------------------------------------------------
// Pre-swizzle Wk/Wv [50,1024] f32 into bf16 WMMA B-fragment order.
// Fragment (kb, nt): 32 lanes x 16 bf16; lane&15 = N, lane>>4 = K-half;
// element j covers K = j + 8*half + (j>=8 ? 8 : 0) within the 32-wide K block.
// Lane data contiguous (32 B) -> one aligned v16bf load in the GEMM.
// ---------------------------------------------------------------------------
__global__ __launch_bounds__(256)
void prep_wfrag_kernel(const float* __restrict__ Wk, const float* __restrict__ Wv,
                       __bf16* __restrict__ wkf, __bf16* __restrict__ wvf) {
    int tid  = blockIdx.x * 256 + threadIdx.x;   // 131072 total
    int mat  = tid >> 16;                        // 0 = Wk, 1 = Wv
    int r    = tid & 65535;
    int j    = r & 15;
    int lane = (r >> 4) & 31;
    int frag = r >> 9;                           // kb*4 + nt, 0..127
    int nt   = frag & 3;
    int kb   = frag >> 2;
    int n    = nt * 16 + (lane & 15);
    int half = lane >> 4;
    int K    = j + half * 8 + ((j >= 8) ? 8 : 0);
    int k    = kb * 32 + K;

    const float* src = mat ? Wv : Wk;
    __bf16*      dst = mat ? wvf : wkf;
    float val = (n < D_) ? src[n * H2_ + k] : 0.0f;
    dst[(size_t)frag * 512 + lane * 16 + j] = (__bf16)val;
}

// ---------------------------------------------------------------------------
// q = selu(h @ Wq^T), padded to [128, 64] f32 (cols >= 50 are zero).
// ---------------------------------------------------------------------------
__global__ __launch_bounds__(64)
void q_proj_kernel(const float* __restrict__ h, const float* __restrict__ Wq,
                   float* __restrict__ qws) {
    __shared__ float hrow[H_];
    const int b = blockIdx.x;
    const int t = threadIdx.x;
    for (int i = t; i < H_; i += 64) hrow[i] = h[b * H_ + i];
    __syncthreads();
    float acc = 0.0f;
    if (t < D_) {
        const float* w = Wq + (size_t)t * H_;
        for (int i = 0; i < H_; ++i) acc += hrow[i] * w[i];
        acc = selu_f(acc);
    }
    qws[b * DP_ + t] = (t < D_) ? acc : 0.0f;
}

// ---------------------------------------------------------------------------
// Fused k/v projection + raw scores. One block per s (128 rows = all b).
// Waves 0-3: K-matrix for 32-row slice (w&3), full q.k score per row.
// Waves 4-7: V-matrix for 32-row slice, selu+store to vws.
// ---------------------------------------------------------------------------
__global__ __launch_bounds__(256)
void kv_proj_score_kernel(const float* __restrict__ ctx,
                          const float* __restrict__ qws,
                          const __bf16* __restrict__ wkf,
                          const __bf16* __restrict__ wvf,
                          float* __restrict__ scores,
                          __bf16* __restrict__ vws) {
    __shared__ __bf16 xsb[128 * XST_];            // 34.8 KB bf16 tile
    const int t    = threadIdx.x;
    const int lane = t & 31;
    const int wave = t >> 5;
    const int rb   = (wave & 3) * 32;             // wave's 32-row slice
    const int isV  = wave >> 2;                   // 0 = K/scores, 1 = V
    const int half = lane >> 4;
    const int n0   = lane & 15;
    const int sIdx = blockIdx.x;                  // block == source position s
    const size_t rowBase = (size_t)sIdx * 128;    // row g = rowBase + m, b = m

    const __bf16* wf = isV ? wvf : wkf;

    v8f acc[2][4] = {};                           // [rowgroup][nt]

    // B-fragment register double buffer; preload kb = 0.
    v16bf bb[2][4];
#pragma unroll
    for (int nt = 0; nt < 4; ++nt)
        bb[0][nt] = *reinterpret_cast<const v16bf*>(
            wf + ((size_t)nt * 32 + lane) * 16);

    for (int kc = 0; kc < 8; ++kc) {              // 8 chunks of 128 K
        __syncthreads();
        {   // stage 128x128 f32 -> bf16 into LDS (cvt folds into store path)
            const int lrow = t >> 1;
            const int ch   = t & 1;
            const float4* src = reinterpret_cast<const float4*>(
                ctx + (rowBase + lrow) * (size_t)H2_ + kc * 128 + ch * 64);
            __bf16* dst = &xsb[lrow * XST_ + ch * 64];
#pragma unroll
            for (int i = 0; i < 8; ++i) {
                float4 a4 = src[2 * i];
                float4 b4 = src[2 * i + 1];
                v8bf p;
                p[0] = (__bf16)a4.x; p[1] = (__bf16)a4.y;
                p[2] = (__bf16)a4.z; p[3] = (__bf16)a4.w;
                p[4] = (__bf16)b4.x; p[5] = (__bf16)b4.y;
                p[6] = (__bf16)b4.z; p[7] = (__bf16)b4.w;
                *reinterpret_cast<v8bf*>(dst + i * 8) = p;
            }
            if (kc + 1 < 8)
                __builtin_prefetch(ctx + (rowBase + lrow) * (size_t)H2_ +
                                   (kc + 1) * 128 + ch * 64, 0, 0);
        }
        __syncthreads();

#pragma unroll
        for (int ks = 0; ks < 4; ++ks) {          // 4 K=32 WMMA steps per chunk
            const int kb  = kc * 4 + ks;
            const int buf = ks & 1;
            // Prefetch next K-step's B fragments into the alternate buffer;
            // overlaps with the WMMA chain below.
            if (kb + 1 < 32) {
#pragma unroll
                for (int nt = 0; nt < 4; ++nt)
                    bb[buf ^ 1][nt] = *reinterpret_cast<const v16bf*>(
                        wf + ((size_t)((kb + 1) * 4 + nt) * 32 + lane) * 16);
            }
            // A fragments straight from bf16 LDS: two contiguous 16B reads
            // per row group (K {0..7,16..23} / {8..15,24..31} per lane-half).
#pragma unroll
            for (int rg = 0; rg < 2; ++rg) {
                const int lrow = rb + rg * 16 + n0;
                const __bf16* ab = &xsb[lrow * XST_ + ks * 32 + half * 8];
                v8bf lo = *reinterpret_cast<const v8bf*>(ab);
                v8bf hi = *reinterpret_cast<const v8bf*>(ab + 16);
                v16bf a = __builtin_shufflevector(
                    lo, hi, 0, 1, 2, 3, 4, 5, 6, 7, 8, 9, 10, 11, 12, 13, 14, 15);
#pragma unroll
                for (int nt = 0; nt < 4; ++nt)
                    acc[rg][nt] = __builtin_amdgcn_wmma_f32_16x16x32_bf16(
                        false, a, false, bb[buf][nt], (short)0,
                        acc[rg][nt], false, false);
            }
        }
    }

    // Epilogue. Accumulator layout: vgpr r, lane -> M = r + 8*(lane>>4), N = lane&15.
    if (isV) {
#pragma unroll
        for (int rg = 0; rg < 2; ++rg)
#pragma unroll
            for (int r = 0; r < 8; ++r) {
                const int   Mt = rb + rg * 16 + r + 8 * half;
                const size_t g = rowBase + Mt;
#pragma unroll
                for (int nt = 0; nt < 4; ++nt)
                    vws[g * DP_ + nt * 16 + n0] = (__bf16)selu_f(acc[rg][nt][r]);
            }
    } else {
#pragma unroll
        for (int rg = 0; rg < 2; ++rg)
#pragma unroll
            for (int r = 0; r < 8; ++r) {
                const int Mt = rb + rg * 16 + r + 8 * half;   // == batch b
                float rowsum = 0.0f;
#pragma unroll
                for (int nt = 0; nt < 4; ++nt)
                    rowsum += selu_f(acc[rg][nt][r]) * qws[Mt * DP_ + nt * 16 + n0];
#pragma unroll
                for (int m = 1; m < 16; m <<= 1)
                    rowsum += __shfl_xor(rowsum, m, 32);
                if (n0 == 0)
                    scores[(size_t)Mt * S_ + sIdx] = rowsum;
            }
    }
}

// ---------------------------------------------------------------------------
// Masked softmax over S per batch row. mask1[b]==0 (mask[b]==0 && max>0)
// => all logits NEG_INF => uniform 1/S.
// ---------------------------------------------------------------------------
__global__ __launch_bounds__(256)
void softmax_kernel(const float* __restrict__ scores,
                    const long long* __restrict__ mask,
                    float* __restrict__ outw) {
    __shared__ float red[256];
    const int b = blockIdx.x;
    const int t = threadIdx.x;

    long long mm = 0;
    for (int i = 0; i < B_; ++i) { long long m = mask[i]; mm = m > mm ? m : mm; }
    const bool masked = (mask[b] <= 0) && (mask[b] < mm);

    float* row = outw + (size_t)b * S_;
    if (masked) {
        const float u = 1.0f / (float)S_;
        for (int s = t; s < S_; s += 256) row[s] = u;
        return;
    }
    const float inv = 0.14142135623730950488f;   // 1/sqrt(50)
    const float* srow = scores + (size_t)b * S_;

    float lmax = -3.4e38f;
    for (int s = t; s < S_; s += 256) lmax = fmaxf(lmax, srow[s] * inv);
    red[t] = lmax; __syncthreads();
    for (int o = 128; o > 0; o >>= 1) {
        if (t < o) red[t] = fmaxf(red[t], red[t + o]);
        __syncthreads();
    }
    const float gmax = red[0];
    __syncthreads();

    float lsum = 0.0f;
    for (int s = t; s < S_; s += 256) lsum += __expf(srow[s] * inv - gmax);
    red[t] = lsum; __syncthreads();
    for (int o = 128; o > 0; o >>= 1) {
        if (t < o) red[t] += red[t + o];
        __syncthreads();
    }
    const float rs = 1.0f / red[0];
    for (int s = t; s < S_; s += 256)
        row[s] = __expf(srow[s] * inv - gmax) * rs;
}

// ---------------------------------------------------------------------------
// c_t[b,d] = sum_s weights[b,s] * v[s,b,d]  (v is bf16, L2-resident)
// ---------------------------------------------------------------------------
__global__ __launch_bounds__(256)
void ctx_out_kernel(const float* __restrict__ outw,
                    const __bf16* __restrict__ vws,
                    float* __restrict__ ct) {
    __shared__ float red[256];
    const int b  = blockIdx.x;
    const int t  = threadIdx.x;
    const int d  = t & 63;
    const int sg = t >> 6;
    const float* wrow = outw + (size_t)b * S_;

    float acc = 0.0f;
    for (int s = sg; s < S_; s += 4) {
        float w = wrow[s];
        float v = (float)vws[((size_t)s * B_ + b) * DP_ + d];
        acc += w * v;
    }
    red[t] = acc;
    __syncthreads();
    if (t < 64) {
        float r = red[t] + red[t + 64] + red[t + 128] + red[t + 192];
        if (d < D_) ct[b * D_ + d] = r;
    }
}

// ---------------------------------------------------------------------------
extern "C" void kernel_launch(void* const* d_in, const int* in_sizes, int n_in,
                              void* d_out, int out_size, void* d_ws, size_t ws_size,
                              hipStream_t stream) {
    (void)in_sizes; (void)n_in; (void)out_size; (void)ws_size;
    // setup_inputs order: h, context, time_step, win, mask, Wq, Wk, Wv
    const float*     h    = (const float*)d_in[0];
    const float*     ctx  = (const float*)d_in[1];
    const long long* mask = (const long long*)d_in[4];
    const float*     Wq   = (const float*)d_in[5];
    const float*     Wk   = (const float*)d_in[6];
    const float*     Wv   = (const float*)d_in[7];

    float* out     = (float*)d_out;
    float* ct      = out;                 // [128, 50]
    float* weights = out + B_ * D_;       // [128, 2048]

    // Workspace layout (all offsets 32B-aligned):
    char*   ws     = (char*)d_ws;
    float*  qws    = (float*)ws;                                  // 32 KB
    float*  scores = (float*)(ws + 32768);                        // 1 MB
    __bf16* wkf    = (__bf16*)(ws + 32768 + 1048576);             // 128 KB
    __bf16* wvf    = wkf + 65536;                                 // 128 KB
    __bf16* vws    = wvf + 65536;                                 // 33.5 MB

    prep_wfrag_kernel   <<<512, 256, 0, stream>>>(Wk, Wv, wkf, wvf);
    q_proj_kernel       <<<B_,   64, 0, stream>>>(h, Wq, qws);
    kv_proj_score_kernel<<<S_,  256, 0, stream>>>(ctx, qws, wkf, wvf, scores, vws);
    softmax_kernel      <<<B_,  256, 0, stream>>>(scores, mask, weights);
    ctx_out_kernel      <<<B_,  256, 0, stream>>>(weights, vws, ct);
}
